// MultilayerGatedGraphConv_23235773071823
// MI455X (gfx1250) — matrix operand order repaired and checked by reading, not compiled, not run on hardware
//
#include <hip/hip_runtime.h>
#include <hip/hip_bf16.h>

typedef __attribute__((ext_vector_type(16))) _Float16 v16h;
typedef __attribute__((ext_vector_type(8)))  _Float16 v8h;
typedef __attribute__((ext_vector_type(8)))  float    v8f;

#define NN 50000
#define DD 128
#define TT 4
#define HH 256
#define LL 3
#define EE 800000
#define GG 512

// ---------------------------------------------------------------- zero (float4)
__global__ void zero_kernel(float* __restrict__ p, long n4) {
    long i      = (long)blockIdx.x * blockDim.x + threadIdx.x;
    long stride = (long)gridDim.x * blockDim.x;
    float4 z = make_float4(0.f, 0.f, 0.f, 0.f);
    for (; i < n4; i += stride) ((float4*)p)[i] = z;
}

// ------------------------------------------------- edge gather + scatter-add (f32 atomics, L2-resident)
__global__ void scatter_kernel(const float* __restrict__ h,
                               const float* __restrict__ ew,
                               const int*   __restrict__ ei,   // [2, E]
                               const int*   __restrict__ et,   // [E]
                               float* __restrict__ agg) {      // [N*T, D]
    long t = (long)blockIdx.x * blockDim.x + threadIdx.x;
    if (t >= (long)EE * 32) return;
    int e = (int)(t >> 5);
    int c = (int)((t & 31) << 2);         // 4 floats per thread
    // prefetch a future gather row into the cache hierarchy (global_prefetch_b8)
    const int PF = 4096;
    if (e + PF < EE) {
        int spf = ei[e + PF];
        __builtin_prefetch(h + (long)spf * DD + c, 0, 0);
    }
    int src = ei[e];
    int dst = ei[EE + e];
    int ty  = et[e];
    float w = ew[e];
    float4 hv = *(const float4*)(h + (long)src * DD + c);
    float* base = agg + ((long)dst * TT + ty) * DD + c;
    unsafeAtomicAdd(base + 0, w * hv.x);
    unsafeAtomicAdd(base + 1, w * hv.y);
    unsafeAtomicAdd(base + 2, w * hv.z);
    unsafeAtomicAdd(base + 3, w * hv.w);
}

// ------------------------------------------------------- WMMA GEMM: Out = act(A[M,K] * W[NC,K]^T + bias)
// BM=128, BN=128, BK=32, 256 threads (8 waves, wave32).
// Waves tiled 4x2; each wave computes 32x64 = 2x4 WMMA 16x16x32 f16 tiles.
// Double-buffered LDS staging: one barrier per K-step, global loads of step k+1
// overlap WMMAs of step k.
template<bool A_HALF, bool RELU, bool OUT_HALF>
__global__ __launch_bounds__(256)
void gemm_wmma_kernel(const void* __restrict__ Avoid, const float* __restrict__ W,
                      const float* __restrict__ bias, void* __restrict__ Ovoid,
                      int M, int K, int NC) {
    __shared__ _Float16 As[2][128 * 32];
    __shared__ _Float16 Bs[2][128 * 32];

    const int tid  = threadIdx.x;
    const int lane = tid & 31;
    const int wave = tid >> 5;
    const int wm   = wave >> 1;     // 0..3  (M sub-tile: 32 rows)
    const int wn   = wave & 1;      // 0..1  (N sub-tile: 64 cols)
    const int lo   = lane & 15;
    const int hi   = lane >> 4;
    const int m0   = blockIdx.x * 128;
    const int n0   = blockIdx.y * 128;

    v8f acc[2][4] = {};

    const int srow = tid >> 1;            // 0..127 (staging row, both A and B)
    const int scol = (tid & 1) * 16;      // 0 or 16
    const int gmA  = m0 + srow;
    const bool aok = (gmA < M);

    v8h ra0, ra1, rb0, rb1;

    auto loadA = [&](int k0) {
        if (aok) {
            if (A_HALF) {
                const _Float16* Ap = (const _Float16*)Avoid + (long)gmA * K + k0 + scol;
                ra0 = *(const v8h*)(Ap);
                ra1 = *(const v8h*)(Ap + 8);
            } else {
                const float* Ap = (const float*)Avoid + (long)gmA * K + k0 + scol;
                float fv[16];
                #pragma unroll
                for (int q = 0; q < 4; ++q) ((float4*)fv)[q] = ((const float4*)Ap)[q];
                #pragma unroll
                for (int u = 0; u < 8; ++u) { ra0[u] = (_Float16)fv[u]; ra1[u] = (_Float16)fv[8 + u]; }
            }
        } else {
            ra0 = (v8h){}; ra1 = (v8h){};
        }
    };
    auto loadB = [&](int k0) {
        const float* Wp = W + (long)(n0 + srow) * K + k0 + scol;
        float fv[16];
        #pragma unroll
        for (int q = 0; q < 4; ++q) ((float4*)fv)[q] = ((const float4*)Wp)[q];
        #pragma unroll
        for (int u = 0; u < 8; ++u) { rb0[u] = (_Float16)fv[u]; rb1[u] = (_Float16)fv[8 + u]; }
    };
    auto stage = [&](int buf) {
        _Float16* da = &As[buf][srow * 32 + scol];
        *(v8h*)(da)     = ra0;
        *(v8h*)(da + 8) = ra1;
        _Float16* db = &Bs[buf][srow * 32 + scol];
        *(v8h*)(db)     = rb0;
        *(v8h*)(db + 8) = rb1;
    };

    const int KT = K / 32;
    loadA(0); loadB(0);
    stage(0);
    __syncthreads();

    for (int kt = 0; kt < KT; ++kt) {
        const int cur = kt & 1;
        const bool more = (kt + 1 < KT);
        if (more) { loadA((kt + 1) * 32); loadB((kt + 1) * 32); }

        // ---- fragments per ISA VGPR layouts ----
        union Frag { v16h v; v8h h[2]; };
        Frag af[2], bf[4];
        #pragma unroll
        for (int i = 0; i < 2; ++i) {
            // A: lane holds row lo, K chunks {kb..kb+7, 16+kb..16+kb+7}, kb = hi*8
            const _Float16* ap = &As[cur][(wm * 32 + i * 16 + lo) * 32 + hi * 8];
            af[i].h[0] = *(const v8h*)(ap);
            af[i].h[1] = *(const v8h*)(ap + 16);
        }
        #pragma unroll
        for (int j = 0; j < 4; ++j) {
            // B: lane holds out-col lo, K chunks {kb..kb+7, 16+kb..16+kb+7}, kb = hi*8
            const _Float16* bp = &Bs[cur][(wn * 64 + j * 16 + lo) * 32 + hi * 8];
            bf[j].h[0] = *(const v8h*)(bp);
            bf[j].h[1] = *(const v8h*)(bp + 16);
        }
        #pragma unroll
        for (int i = 0; i < 2; ++i)
            #pragma unroll
            for (int j = 0; j < 4; ++j)
                acc[i][j] = __builtin_amdgcn_wmma_f32_16x16x32_f16(
                    false, af[i].v, false, bf[j].v, (short)0, acc[i][j], false, false);

        if (more) stage(1 - cur);
        __syncthreads();
    }

    // ---- epilogue: C/D layout -> element v is (M = v + 8*hi, N = lo) ----
    #pragma unroll
    for (int j = 0; j < 4; ++j) {
        int gn = n0 + wn * 64 + j * 16 + lo;
        float b = bias[gn];
        #pragma unroll
        for (int i = 0; i < 2; ++i) {
            #pragma unroll
            for (int v = 0; v < 8; ++v) {
                int gm = m0 + wm * 32 + i * 16 + v + hi * 8;
                if (gm < M) {
                    float val = acc[i][j][v] + b;
                    if (RELU) val = fmaxf(val, 0.f);
                    if (OUT_HALF) ((_Float16*)Ovoid)[(long)gm * NC + gn] = (_Float16)val;
                    else          ((float*)Ovoid)[(long)gm * NC + gn]    = val;
                }
            }
        }
    }
}

// ---------------------------------------------------------------- GRU elementwise
__global__ void gru_kernel(const float* __restrict__ gi, const float* __restrict__ gh,
                           const float* __restrict__ h, float* __restrict__ hout) {
    long t = (long)blockIdx.x * blockDim.x + threadIdx.x;
    if (t >= (long)NN * DD) return;
    int n = (int)(t >> 7);
    int d = (int)(t & 127);
    long b3 = (long)n * 384;
    float ir = gi[b3 + d], iz = gi[b3 + 128 + d], in_ = gi[b3 + 256 + d];
    float hr = gh[b3 + d], hz = gh[b3 + 128 + d], hn  = gh[b3 + 256 + d];
    float r  = 1.f / (1.f + __expf(-(ir + hr)));
    float z  = 1.f / (1.f + __expf(-(iz + hz)));
    float nn = tanhf(in_ + r * hn);
    hout[t] = (1.f - z) * nn + z * h[t];
}

// ---------------------------------------------------------------- graph readout + h copy
__global__ void readout_kernel(const float* __restrict__ h, const int* __restrict__ n2g,
                               float* __restrict__ gf, float* __restrict__ hcopy) {
    long t = (long)blockIdx.x * blockDim.x + threadIdx.x;
    if (t >= (long)NN * DD) return;
    int n = (int)(t >> 7);
    int d = (int)(t & 127);
    float v = h[t];
    hcopy[t] = v;
    unsafeAtomicAdd(&gf[(long)n2g[n] * DD + d], v);
}

extern "C" void kernel_launch(void* const* d_in, const int* in_sizes, int n_in,
                              void* d_out, int out_size, void* d_ws, size_t ws_size,
                              hipStream_t stream) {
    const float* node_feature = (const float*)d_in[0];
    const float* edge_weight  = (const float*)d_in[1];
    const float* mlp_w        = (const float*)d_in[2];   // [L, H, T*D]
    const float* mlp_b        = (const float*)d_in[3];   // [L, H]
    const float* w_ih         = (const float*)d_in[4];   // [L, 3D, H]
    const float* w_hh         = (const float*)d_in[5];   // [L, 3D, D]
    const float* b_ih         = (const float*)d_in[6];   // [L, 3D]
    const float* b_hh         = (const float*)d_in[7];   // [L, 3D]
    const int*   edge_index   = (const int*)d_in[8];     // [2, E]
    const int*   edge_type    = (const int*)d_in[9];     // [E]
    const int*   node2graph   = (const int*)d_in[10];    // [N]

    char* ws = (char*)d_ws;
    size_t off = 0;
    auto alloc = [&](size_t bytes) -> void* {
        void* p = ws + off;
        off += (bytes + 255) & ~(size_t)255;
        return p;
    };
    float*    agg = (float*)alloc((size_t)NN * TT * DD * sizeof(float));   // 102.4 MB
    _Float16* hid = (_Float16*)alloc((size_t)NN * HH * sizeof(_Float16));  //  25.6 MB
    float*    gi  = (float*)alloc((size_t)NN * 384 * sizeof(float));       //  76.8 MB
    float*    gh  = (float*)alloc((size_t)NN * 384 * sizeof(float));       //  76.8 MB
    float*    hb0 = (float*)alloc((size_t)NN * DD * sizeof(float));        //  25.6 MB
    float*    hb1 = (float*)alloc((size_t)NN * DD * sizeof(float));        //  25.6 MB
    (void)ws_size;

    const int MT = (NN + 127) / 128;   // 391 M-tiles
    const float* hcur = node_feature;
    float* hbufs[2] = {hb0, hb1};

    for (int l = 0; l < LL; ++l) {
        // 1) zero agg, scatter messages
        zero_kernel<<<2048, 256, 0, stream>>>(agg, (long)NN * TT * DD / 4);
        long st = (long)EE * 32;
        scatter_kernel<<<(int)((st + 255) / 256), 256, 0, stream>>>(
            hcur, edge_weight, edge_index, edge_type, agg);

        // 2) hid = relu(agg[N,512] @ mlp_w[l]^T + mlp_b[l])  -> f16
        gemm_wmma_kernel<false, true, true><<<dim3(MT, HH / 128), 256, 0, stream>>>(
            agg, mlp_w + (size_t)l * HH * TT * DD, mlp_b + (size_t)l * HH,
            hid, NN, TT * DD, HH);

        // 3) gi = hid[N,256] @ w_ih[l]^T + b_ih[l]           -> f32
        gemm_wmma_kernel<true, false, false><<<dim3(MT, 384 / 128), 256, 0, stream>>>(
            hid, w_ih + (size_t)l * 384 * HH, b_ih + (size_t)l * 384,
            gi, NN, HH, 384);

        // 4) gh = h[N,128] @ w_hh[l]^T + b_hh[l]             -> f32
        gemm_wmma_kernel<false, false, false><<<dim3(MT, 384 / 128), 256, 0, stream>>>(
            (const void*)hcur, w_hh + (size_t)l * 384 * DD, b_hh + (size_t)l * 384,
            gh, NN, DD, 384);

        // 5) GRU update
        long gn = (long)NN * DD;
        gru_kernel<<<(int)((gn + 255) / 256), 256, 0, stream>>>(gi, gh, hcur, hbufs[l & 1]);
        hcur = hbufs[l & 1];
    }

    // readout: d_out = [graph_feature (G*D) | h (N*D)]
    float* gf   = (float*)d_out;
    float* hout = (float*)d_out + (size_t)GG * DD;
    zero_kernel<<<64, 256, 0, stream>>>(gf, (long)GG * DD / 4);
    long gn = (long)NN * DD;
    readout_kernel<<<(int)((gn + 255) / 256), 256, 0, stream>>>(hcur, node2graph, gf, hout);
}
